// DigitDecoder_81681688035681
// MI455X (gfx1250) — matrix operand order repaired and checked by reading, not compiled, not run on hardware
//
#include <hip/hip_runtime.h>
#include <hip/hip_bf16.h>
#include <math.h>

#define Dm   384
#define Hh   12
#define HDim 32
#define Pp   576
#define Bn   256
#define FFn  1536
#define SCALE_QK 0.17677669529663689f  // 1/sqrt(32)

typedef __attribute__((ext_vector_type(16))) __bf16       v16bf;
typedef __attribute__((ext_vector_type(8)))  float        v8f;
typedef __attribute__((ext_vector_type(4)))  unsigned int uint4v;

union FragB {
    v16bf v;
    struct { uint4v lo, hi; } u;
};
union PackBF {
    __bf16 h[2];
    unsigned int u;
};

__device__ __forceinline__ float wave_max(float v) {
    for (int o = 16; o > 0; o >>= 1) v = fmaxf(v, __shfl_xor(v, o, 32));
    return v;
}
__device__ __forceinline__ float wave_sum(float v) {
    for (int o = 16; o > 0; o >>= 1) v += __shfl_xor(v, o, 32);
    return v;
}
__device__ __forceinline__ float dotn(const float* a, const float* b, int n) {
    float s = 0.f;
#pragma unroll 4
    for (int k = 0; k < n; ++k) s = fmaf(a[k], b[k], s);
    return s;
}

// LayerNorm over 2 rows of 384 held in LDS. All 256 threads must enter.
__device__ void ln2rows(const float* in, const float* g, const float* b,
                        float* out, float* red) {
    const int tid = threadIdx.x, wave = tid >> 5, lane = tid & 31;
    if (wave < 2) {
        float s = 0.f, s2 = 0.f;
        for (int i = lane; i < Dm; i += 32) {
            float v = in[wave * Dm + i];
            s += v; s2 += v * v;
        }
        s = wave_sum(s); s2 = wave_sum(s2);
        if (lane == 0) {
            float mu  = s / (float)Dm;
            float var = s2 / (float)Dm - mu * mu;
            red[wave * 2 + 0] = mu;
            red[wave * 2 + 1] = rsqrtf(var + 1e-5f);
        }
    }
    __syncthreads();
    for (int e = tid; e < 2 * Dm; e += 256) {
        int r = e / Dm, d = e % Dm;
        out[e] = (in[e] - red[r * 2]) * red[r * 2 + 1] * g[d] + b[d];
    }
    __syncthreads();
}

// ---------------------------------------------------------------------------
// Kernel 0: batch-independent query-stream prefix.
// Produces ws_tgt (2x384 f32, residual after self-attention) and
// ws_qp (32x384 bf16; rows 0..23 = scale * q_ca[head] @ Wk_ca[head], rows
// 24..31 zero padding).
// ---------------------------------------------------------------------------
__global__ __launch_bounds__(256) void k0_precompute(
    const float* pos_q, const float* sa_w_in, const float* sa_b_in,
    const float* sa_w_out, const float* sa_b_out,
    const float* ca_w_in, const float* ca_b_in,
    const float* nq_g, const float* nq_b, const float* nc_g, const float* nc_b,
    const float* n1_g, const float* n1_b,
    float* ws_tgt, __bf16* ws_qp) {
    __shared__ float q0s[768], qns[768], cns[768];
    __shared__ float qhs[768], khs[768], vhs[768], os[768];
    __shared__ float tgts[768], x1s[768], qcs[768];
    __shared__ float red[8];
    const int tid = threadIdx.x;

    for (int e = tid; e < 768; e += 256) q0s[e] = pos_q[e];
    __syncthreads();

    ln2rows(q0s, nq_g, nq_b, qns, red);
    ln2rows(q0s, nc_g, nc_b, cns, red);

    // self-attn QKV projections (2x384 each)
    for (int e = tid; e < 768; e += 256) {
        int r = e / Dm, d = e % Dm;
        qhs[e] = sa_b_in[d]          + dotn(qns + r * Dm, sa_w_in + (size_t)d * Dm, Dm);
        khs[e] = sa_b_in[Dm + d]     + dotn(cns + r * Dm, sa_w_in + (size_t)(Dm + d) * Dm, Dm);
        vhs[e] = sa_b_in[2 * Dm + d] + dotn(cns + r * Dm, sa_w_in + (size_t)(2 * Dm + d) * Dm, Dm);
    }
    __syncthreads();

    // per-(query, head) 2-key attention
    if (tid < 24) {
        int q = tid / Hh, h = tid % Hh, base = h * HDim;
        float s0 = 0.f, s1 = 0.f;
        for (int j = 0; j < HDim; ++j) {
            float qv = qhs[q * Dm + base + j];
            s0 += qv * khs[0 * Dm + base + j];
            s1 += qv * khs[1 * Dm + base + j];
        }
        s0 *= SCALE_QK; s1 *= SCALE_QK;
        float mx = fmaxf(s0, s1);
        float e0 = __expf(s0 - mx), e1 = __expf(s1 - mx);
        float a0 = e0 / (e0 + e1), a1 = e1 / (e0 + e1);
        for (int j = 0; j < HDim; ++j)
            os[q * Dm + base + j] = a0 * vhs[base + j] + a1 * vhs[Dm + base + j];
    }
    __syncthreads();

    // sa output projection + residual -> tgt
    for (int e = tid; e < 768; e += 256) {
        int r = e / Dm, d = e % Dm;
        float v = q0s[e] + sa_b_out[d] + dotn(os + r * Dm, sa_w_out + (size_t)d * Dm, Dm);
        tgts[e] = v;
        ws_tgt[e] = v;
    }
    __syncthreads();

    ln2rows(tgts, n1_g, n1_b, x1s, red);

    // cross-attn query projection (2x384)
    for (int e = tid; e < 768; e += 256) {
        int r = e / Dm, d = e % Dm;
        qcs[e] = ca_b_in[d] + dotn(x1s + r * Dm, ca_w_in + (size_t)d * Dm, Dm);
    }
    __syncthreads();

    // Qp[qh][k] = scale * sum_j qc[q][h*32+j] * Wk_ca[h*32+j][k]
    for (int e = tid; e < 32 * Dm; e += 256) {
        int row = e / Dm, k = e % Dm;
        float v = 0.f;
        if (row < 24) {
            int q = row / Hh, h = row % Hh;
            float s = 0.f;
            for (int j = 0; j < HDim; ++j)
                s += qcs[q * Dm + h * HDim + j] *
                     ca_w_in[(size_t)(Dm + h * HDim + j) * Dm + k];
            v = s * SCALE_QK;
        }
        ws_qp[e] = (__bf16)v;
    }
}

// ---------------------------------------------------------------------------
// Kernel 1: per-batch fused cross-attention + epilogue.
// LDS plan (63488 B static, phase-overlapped):
//   [0, 36864)      : Sbf 32x576 bf16 (passes 1-2), then y 24x384 f32
//   [36864, 61440)  : xT 384x32 bf16 (pass 2), then epilogue f32 buffers
// ---------------------------------------------------------------------------
__global__ __launch_bounds__(256) void k1_main(
    const float* __restrict__ X, const __bf16* __restrict__ qp,
    const float* __restrict__ ws_tgt,
    const float* ca_w_in, const float* ca_b_in,
    const float* ca_w_out, const float* ca_b_out,
    const float* w1, const float* b1, const float* w2, const float* b2,
    const float* n2_g, const float* n2_b, const float* nf_g, const float* nf_b,
    const float* head_w, const float* head_b,
    float* out_all, float* out_pos) {
    __shared__ __align__(64) unsigned char smem[63488];
    __bf16* Sbf = (__bf16*)smem;                 // 32 x 576
    float*  y   = (float*)smem;                  // 24 x 384 (after pass 2)
    __bf16* xT  = (__bf16*)(smem + 36864);       // 384 x 32
    float*  epi = (float*)(smem + 36864);

    const int tid  = threadIdx.x;
    const int wave = tid >> 5, lane = tid & 31;
    const int m  = lane & 15;
    const int hi = lane >> 4;                    // 0/1 lane half
    const int b  = blockIdx.x;
    const float* Xb = X + (size_t)b * Pp * Dm;

    // ---- Pass 1: scores S = X @ Qp^T  (576 x 24), straight to LDS bf16 ----
    for (int rt = wave; rt < Pp / 16; rt += 8) {
        const int t0 = rt * 16;
        const float* xrow = Xb + (size_t)(t0 + m) * Dm;
        v8f acc0 = {0, 0, 0, 0, 0, 0, 0, 0};
        v8f acc1 = {0, 0, 0, 0, 0, 0, 0, 0};
        __builtin_prefetch(Xb + (size_t)(t0 + 16 + m) * Dm, 0, 1);
#pragma unroll 4
        for (int kc = 0; kc < 12; ++kc) {
            const int k0 = kc * 32, kb = hi * 8;
            const float* pA = xrow + k0 + kb;
            float4 f0 = *(const float4*)(pA);
            float4 f1 = *(const float4*)(pA + 4);
            float4 f2 = *(const float4*)(pA + 16);
            float4 f3 = *(const float4*)(pA + 20);
            FragB A;
            A.v[0]  = (__bf16)f0.x; A.v[1]  = (__bf16)f0.y;
            A.v[2]  = (__bf16)f0.z; A.v[3]  = (__bf16)f0.w;
            A.v[4]  = (__bf16)f1.x; A.v[5]  = (__bf16)f1.y;
            A.v[6]  = (__bf16)f1.z; A.v[7]  = (__bf16)f1.w;
            A.v[8]  = (__bf16)f2.x; A.v[9]  = (__bf16)f2.y;
            A.v[10] = (__bf16)f2.z; A.v[11] = (__bf16)f2.w;
            A.v[12] = (__bf16)f3.x; A.v[13] = (__bf16)f3.y;
            A.v[14] = (__bf16)f3.z; A.v[15] = (__bf16)f3.w;
            FragB B0, B1;
            const __bf16* pb0 = qp + (size_t)m * Dm + k0 + hi * 8;
            const __bf16* pb1 = qp + (size_t)(16 + m) * Dm + k0 + hi * 8;
            B0.u.lo = *(const uint4v*)pb0; B0.u.hi = *(const uint4v*)(pb0 + 16);
            B1.u.lo = *(const uint4v*)pb1; B1.u.hi = *(const uint4v*)(pb1 + 16);
            acc0 = __builtin_amdgcn_wmma_f32_16x16x32_bf16(
                false, A.v, false, B0.v, (short)0, acc0, false, false);
            acc1 = __builtin_amdgcn_wmma_f32_16x16x32_bf16(
                false, A.v, false, B1.v, (short)0, acc1, false, false);
        }
        // D tile: M = r + 8*hi (token), N = m (qh col).  Scores stored [qh][tok].
#pragma unroll
        for (int r = 0; r < 8; ++r) {
            int tok = t0 + r + 8 * hi;
            Sbf[(size_t)m * Pp + tok]        = (__bf16)acc0[r];
            Sbf[(size_t)(16 + m) * Pp + tok] = (__bf16)acc1[r];
        }
    }
    __syncthreads();

    // ---- Softmax over 576 keys, rows 0..23 (rows 24..31 remain zero) ----
    for (int row = wave; row < 24; row += 8) {
        float vals[18];
        float mx = -1e30f;
#pragma unroll
        for (int i = 0; i < 18; ++i) {
            vals[i] = (float)Sbf[(size_t)row * Pp + lane + 32 * i];
            mx = fmaxf(mx, vals[i]);
        }
        mx = wave_max(mx);
        float s = 0.f;
#pragma unroll
        for (int i = 0; i < 18; ++i) { vals[i] = __expf(vals[i] - mx); s += vals[i]; }
        s = wave_sum(s);
        const float inv = 1.f / s;
#pragma unroll
        for (int i = 0; i < 18; ++i)
            Sbf[(size_t)row * Pp + lane + 32 * i] = (__bf16)(vals[i] * inv);
    }
    __syncthreads();

    // ---- Pass 2: y = attn(24x576) @ X(576x384), staged transposed tiles ----
    const int mt = wave & 1;         // output row-tile (qh 0..15 / 16..31)
    const int ng = wave >> 1;        // 6 column-tiles per wave
    v8f acc[6];
#pragma unroll
    for (int j = 0; j < 6; ++j) acc[j] = (v8f){0, 0, 0, 0, 0, 0, 0, 0};

    for (int kt = 0; kt < Pp / 32; ++kt) {
        // Stage 32 tokens x 384 features, transposed to bf16 xT[f][t].
        // Vectorized: token-pair (fast across lanes) x float4 feature group.
        for (int it = tid; it < 16 * 96; it += 256) {
            const int tp = it & 15;          // token pair index (tokens 2tp, 2tp+1)
            const int f4 = it >> 4;          // feature group of 4
            const float* base = Xb + (size_t)(kt * 32 + 2 * tp) * Dm + f4 * 4;
            const float4 a = *(const float4*)base;
            const float4 c = *(const float4*)(base + Dm);
            PackBF p0, p1, p2, p3;
            p0.h[0] = (__bf16)a.x; p0.h[1] = (__bf16)c.x;
            p1.h[0] = (__bf16)a.y; p1.h[1] = (__bf16)c.y;
            p2.h[0] = (__bf16)a.z; p2.h[1] = (__bf16)c.z;
            p3.h[0] = (__bf16)a.w; p3.h[1] = (__bf16)c.w;
            unsigned int* dst = (unsigned int*)(xT + (size_t)(f4 * 4) * 32 + 2 * tp);
            dst[0]  = p0.u;                  // feature f4*4+0, tokens 2tp..2tp+1
            dst[16] = p1.u;                  // +1 feature row = 32 bf16 = 16 dwords
            dst[32] = p2.u;
            dst[48] = p3.u;
        }
        __syncthreads();
        FragB A;
        {
            const __bf16* pa = Sbf + (size_t)(mt * 16 + m) * Pp + kt * 32 + hi * 8;
            A.u.lo = *(const uint4v*)pa;
            A.u.hi = *(const uint4v*)(pa + 16);
        }
#pragma unroll
        for (int j = 0; j < 6; ++j) {
            const int n0 = (ng * 6 + j) * 16;
            FragB Bf;
            const __bf16* pb = xT + (size_t)(n0 + m) * 32 + hi * 8;
            Bf.u.lo = *(const uint4v*)pb;
            Bf.u.hi = *(const uint4v*)(pb + 16);
            acc[j] = __builtin_amdgcn_wmma_f32_16x16x32_bf16(
                false, A.v, false, Bf.v, (short)0, acc[j], false, false);
        }
        __syncthreads();
    }
    // write y (rows 0..23 valid), overwrites Sbf region
#pragma unroll
    for (int j = 0; j < 6; ++j) {
        const int n0 = (ng * 6 + j) * 16;
#pragma unroll
        for (int r = 0; r < 8; ++r) {
            int row = mt * 16 + r + 8 * hi;
            if (row < 24) y[(size_t)row * Dm + n0 + m] = acc[j][r];
        }
    }
    __syncthreads();

    // ---- Epilogue (all f32 VALU) ----
    float* ovec = epi;               // 768
    float* tvec = epi + 768;         // 768
    float* xln  = epi + 1536;        // 768
    float* h1v  = epi + 2304;        // 3072
    float* pl   = epi + 5376;        // 22
    float* red  = epi + 5400;        // 8

    // V-projection of attention-weighted inputs + bv
    for (int e = tid; e < 768; e += 256) {
        int q = e / Dm, d = e % Dm, h = d >> 5;
        ovec[e] = ca_b_in[2 * Dm + d] +
                  dotn(y + (size_t)(q * Hh + h) * Dm,
                       ca_w_in + (size_t)(2 * Dm + d) * Dm, Dm);
    }
    __syncthreads();

    // CA output projection + residual
    for (int e = tid; e < 768; e += 256) {
        int q = e / Dm, d = e % Dm;
        tvec[e] = ws_tgt[e] + ca_b_out[d] +
                  dotn(ovec + q * Dm, ca_w_out + (size_t)d * Dm, Dm);
    }
    __syncthreads();

    ln2rows(tvec, n2_g, n2_b, xln, red);

    // MLP up + exact GELU
    for (int e = tid; e < 2 * FFn; e += 256) {
        int q = e / FFn, f = e % FFn;
        float v = b1[f] + dotn(xln + q * Dm, w1 + (size_t)f * Dm, Dm);
        h1v[e] = 0.5f * v * (1.f + erff(v * 0.70710678118654752f));
    }
    __syncthreads();

    // MLP down + residual
    for (int e = tid; e < 768; e += 256) {
        int q = e / Dm, d = e % Dm;
        tvec[e] += b2[d] + dotn(h1v + q * FFn, w2 + (size_t)d * FFn, FFn);
    }
    __syncthreads();

    ln2rows(tvec, nf_g, nf_b, xln, red);   // reuse xln as final query

    if (tid < 22) {
        int q = tid / 11, c = tid % 11;
        pl[q * 11 + c] = head_b[c] + dotn(xln + q * Dm, head_w + (size_t)c * Dm, Dm);
    }
    __syncthreads();

    float* oa = out_all + (size_t)b * 101;
    for (int i = tid; i < 101; i += 256) {
        float v;
        if (i < 10)       v = pl[i];
        else if (i < 100) v = pl[i / 10] + pl[11 + i % 10];
        else              v = 0.5f * (pl[10] + pl[21]);
        oa[i] = v;
    }
    for (int i = tid; i < 22; i += 256)
        out_pos[(size_t)b * 22 + i] = pl[i];
}

extern "C" void kernel_launch(void* const* d_in, const int* in_sizes, int n_in,
                              void* d_out, int out_size, void* d_ws, size_t ws_size,
                              hipStream_t stream) {
    const float* patch    = (const float*)d_in[0];
    const float* pos_q    = (const float*)d_in[1];
    const float* sa_w_in  = (const float*)d_in[2];
    const float* sa_b_in  = (const float*)d_in[3];
    const float* sa_w_out = (const float*)d_in[4];
    const float* sa_b_out = (const float*)d_in[5];
    const float* ca_w_in  = (const float*)d_in[6];
    const float* ca_b_in  = (const float*)d_in[7];
    const float* ca_w_out = (const float*)d_in[8];
    const float* ca_b_out = (const float*)d_in[9];
    const float* w1       = (const float*)d_in[10];
    const float* b1       = (const float*)d_in[11];
    const float* w2       = (const float*)d_in[12];
    const float* b2       = (const float*)d_in[13];
    const float* n1_g     = (const float*)d_in[14];
    const float* n1_b     = (const float*)d_in[15];
    const float* n2_g     = (const float*)d_in[16];
    const float* n2_b     = (const float*)d_in[17];
    const float* nq_g     = (const float*)d_in[18];
    const float* nq_b     = (const float*)d_in[19];
    const float* nc_g     = (const float*)d_in[20];
    const float* nc_b     = (const float*)d_in[21];
    const float* nf_g     = (const float*)d_in[22];
    const float* nf_b     = (const float*)d_in[23];
    const float* head_w   = (const float*)d_in[24];
    const float* head_b   = (const float*)d_in[25];

    float*  ws_tgt = (float*)d_ws;                       // 768 f32
    __bf16* ws_qp  = (__bf16*)((char*)d_ws + 4096);      // 32x384 bf16

    float* out_all = (float*)d_out;                      // B x 101
    float* out_pos = out_all + (size_t)Bn * 101;         // B x 2 x 11

    k0_precompute<<<1, 256, 0, stream>>>(
        pos_q, sa_w_in, sa_b_in, sa_w_out, sa_b_out, ca_w_in, ca_b_in,
        nq_g, nq_b, nc_g, nc_b, n1_g, n1_b, ws_tgt, ws_qp);

    k1_main<<<Bn, 256, 0, stream>>>(
        patch, ws_qp, ws_tgt, ca_w_in, ca_b_in, ca_w_out, ca_b_out,
        w1, b1, w2, b2, n2_g, n2_b, nf_g, nf_b, head_w, head_b,
        out_all, out_pos);

    (void)in_sizes; (void)n_in; (void)out_size; (void)ws_size;
}
